// transform_gcn_86852828659765
// MI455X (gfx1250) — matrix operand order repaired and checked by reading, not compiled, not run on hardware
//
#include <hip/hip_runtime.h>
#include <hip/hip_bf16.h>

// ---------- problem constants (from reference setup_inputs) ----------
#define NB   16      // batch
#define CI   128     // in channels
#define CO   256     // out channels
#define TT   128     // time
#define VV   25      // joints
#define WW_  11      // parts
#define KK_  3       // kernel partitions
#define GG   8       // groups
#define MM   (CO*WW_)        // 2816  (GEMM M, divisible by 128)
#define KD   (CI*VV)         // 3200  (GEMM K, divisible by 64)
#define PP   (NB*TT)         // 2048  (GEMM N, divisible by 128)
#define EPS  1e-5f

typedef __attribute__((ext_vector_type(8)))  float  v8f;
typedef __attribute__((ext_vector_type(8)))  __bf16 bf16x8;
typedef __attribute__((ext_vector_type(16))) __bf16 bf16x16;
typedef __attribute__((__vector_size__(4 * sizeof(int)))) int v4i;

// CDNA5 async global->LDS path (gated: falls back to sync loads if absent)
#if defined(__AMDGCN__) && __has_builtin(__builtin_amdgcn_global_load_async_to_lds_b128) && __has_builtin(__builtin_amdgcn_s_wait_asynccnt)
#define USE_ASYNC_LDS 1
#else
#define USE_ASYNC_LDS 0
#endif

// ---------- prep 1: fused weight tensor  WW[m=c*11+w][kk=i*25+v] (bf16) ----------
__global__ __launch_bounds__(256) void prep_ww_kernel(
    const float* __restrict__ conv_w, const float* __restrict__ down_w,
    const float* __restrict__ A_adj,  const float* __restrict__ resg,
    const float* __restrict__ dbn_g,  const float* __restrict__ dbn_v,
    const float* __restrict__ bn_g,   const float* __restrict__ bn_v,
    __bf16* __restrict__ ww)
{
  int e = blockIdx.x * 256 + threadIdx.x;
  if (e >= MM * KD) return;
  int m  = e / KD, kk = e % KD;
  int c  = m / WW_, w = m - c * WW_;
  int i  = kk / VV, v = kk - i * VV;
  int g  = c & (GG - 1);
  float bs  = bn_g[c] * rsqrtf(bn_v[c] + EPS);
  float acc = 0.f, racc = 0.f;
#pragma unroll
  for (int k = 0; k < KK_; ++k) {
    int o    = k * CO + c;
    int gidx = ((k * GG + g) * VV + v) * WW_ + w;
    acc  += conv_w[o * CI + i] * A_adj[gidx];
    float ds = dbn_g[o] * rsqrtf(dbn_v[o] + EPS);
    racc += ds * down_w[o * CI + i] * resg[gidx];
  }
  ww[(size_t)m * KD + kk] = (__bf16)(bs * acc + racc);
}

// ---------- prep 2: fused bias[m] (fp32) ----------
__global__ __launch_bounds__(256) void prep_bias_kernel(
    const float* __restrict__ conv_b, const float* __restrict__ down_b,
    const float* __restrict__ A_adj,  const float* __restrict__ resg,
    const float* __restrict__ dbn_g,  const float* __restrict__ dbn_b,
    const float* __restrict__ dbn_m,  const float* __restrict__ dbn_v,
    const float* __restrict__ bn_g,   const float* __restrict__ bn_b,
    const float* __restrict__ bn_m,   const float* __restrict__ bn_v,
    float* __restrict__ bias)
{
  int m = blockIdx.x * 256 + threadIdx.x;
  if (m >= MM) return;
  int c = m / WW_, w = m - c * WW_;
  int g = c & (GG - 1);
  float bs = bn_g[c] * rsqrtf(bn_v[c] + EPS);
  float bt = bn_b[c] - bn_m[c] * bs;
  float acc = bt;
#pragma unroll
  for (int k = 0; k < KK_; ++k) {
    int o = k * CO + c;
    float ds = dbn_g[o] * rsqrtf(dbn_v[o] + EPS);
    float dt = dbn_b[o] - dbn_m[o] * ds;
    float rb = ds * down_b[o] + dt;
    float cb = bs * conv_b[o];
    for (int v = 0; v < VV; ++v) {
      int gidx = ((k * GG + g) * VV + v) * WW_ + w;
      acc += cb * A_adj[gidx] + rb * resg[gidx];
    }
  }
  bias[m] = acc;
}

// ---------- prep 3: x (N,Ci,T,V) fp32 -> Xb[p=n*T+t][kk=i*25+v] bf16 ----------
__global__ __launch_bounds__(256) void prep_x_kernel(
    const float* __restrict__ x, __bf16* __restrict__ xb)
{
  int e = blockIdx.x * 256 + threadIdx.x;
  if (e >= PP * KD) return;
  int p  = e / KD, kk = e - p * KD;
  int n  = p >> 7, t = p & (TT - 1);
  int i  = kk / VV, v = kk - i * VV;
  xb[e] = (__bf16)x[(size_t)n * (CI * TT * VV) + (size_t)i * (TT * VV) + t * VV + v];
}

// ---------- GEMM: out = relu(WW(2816x3200) * Xb^T(3200x2048) + bias) ----------
// block tile 128(M) x 128(P); 8 wave32 waves as 4(M) x 2(P); wave = 32x64
// = 2x4 accumulators of v_wmma_f32_16x16x32_bf16.
// K staged in chunks of 64 through double-buffered LDS filled by
// global_load_async_to_lds_b128 (ASYNCcnt-tracked) when available.
#define KC 64    // K elems per stage
#define LP 72    // LDS row pitch in bf16 elems = 144B (16B aligned, bank-conflict-free)

__global__ __launch_bounds__(256) void gcn_gemm_kernel(
    const __bf16* __restrict__ ww, const __bf16* __restrict__ xb,
    const float* __restrict__ bias, float* __restrict__ out)
{
  __shared__ __bf16 As[2][128 * LP];
  __shared__ __bf16 Bs[2][128 * LP];

  const int tid  = threadIdx.x;
  const int lane = tid & 31;
  const int wid  = tid >> 5;
  const int wm   = wid & 3;   // wave row (M)
  const int wp   = wid >> 2;  // wave col (P)
  const int lm   = lane & 15;
  const int hf   = lane >> 4;

  const int m0 = blockIdx.x * 128;
  const int p0 = blockIdx.y * 128;

  v8f acc[2][4];
#pragma unroll
  for (int i = 0; i < 2; ++i)
#pragma unroll
    for (int j = 0; j < 4; ++j)
      acc[i][j] = (v8f){0.f, 0.f, 0.f, 0.f, 0.f, 0.f, 0.f, 0.f};

  // stage a 128x64 bf16 tile of A and B into LDS buffer b
  auto issue_stage = [&](int kc, int b) {
#pragma unroll
    for (int jj = 0; jj < 4; ++jj) {
      int e   = tid + jj * 256;    // 0..1023 : 8 x 16B segments per 128B row
      int row = e >> 3;
      int seg = e & 7;
      const __bf16* ga = &ww[(size_t)(m0 + row) * KD + kc + seg * 8];
      const __bf16* gb = &xb[(size_t)(p0 + row) * KD + kc + seg * 8];
      __bf16* la = &As[b][row * LP + seg * 8];
      __bf16* lb = &Bs[b][row * LP + seg * 8];
#if USE_ASYNC_LDS
      __builtin_amdgcn_global_load_async_to_lds_b128(
          (__attribute__((address_space(1))) v4i*)ga,
          (__attribute__((address_space(3))) v4i*)la, 0, 0);
      __builtin_amdgcn_global_load_async_to_lds_b128(
          (__attribute__((address_space(1))) v4i*)gb,
          (__attribute__((address_space(3))) v4i*)lb, 0, 0);
#else
      *(bf16x8*)la = *(const bf16x8*)ga;
      *(bf16x8*)lb = *(const bf16x8*)gb;
#endif
    }
  };

  issue_stage(0, 0);               // prologue: fill buffer 0

  const int NSTEP = KD / KC;       // 50
  for (int s = 0; s < NSTEP; ++s) {
    const int b = s & 1;
    if (s + 1 < NSTEP) {
      issue_stage((s + 1) * KC, b ^ 1);   // overlap next stage with compute
#if USE_ASYNC_LDS
      __builtin_amdgcn_s_wait_asynccnt(8); // in-order: first 8 (this stage) done
#endif
      if ((s + 2) * KC < KD) {             // gfx1250 global_prefetch_b8, stage+2
        __builtin_prefetch(&ww[(size_t)(m0 + (tid >> 1)) * KD + (s + 2) * KC], 0, 1);
        __builtin_prefetch(&xb[(size_t)(p0 + (tid >> 1)) * KD + (s + 2) * KC], 0, 1);
      }
    } else {
#if USE_ASYNC_LDS
      __builtin_amdgcn_s_wait_asynccnt(0);
#endif
    }
    __syncthreads();                 // stage b visible to all waves

#pragma unroll
    for (int ks = 0; ks < 2; ++ks) { // two 32-wide WMMA k-steps per stage
      // A frags: 16x32 bf16, lane(l%16)=M row, hf selects K-halves per ISA layout
      bf16x16 afr[2];
#pragma unroll
      for (int i = 0; i < 2; ++i) {
        int r = wm * 32 + i * 16 + lm;
        union { bf16x16 v; bf16x8 h[2]; } u;
        u.h[0] = *(const bf16x8*)&As[b][r * LP + ks * 32 + hf * 8];
        u.h[1] = *(const bf16x8*)&As[b][r * LP + ks * 32 + hf * 8 + 16];
        afr[i] = u.v;
      }
      // B frags: 32x16 bf16, lane: N=l%16, K = hf*8..hf*8+7 and 16+hf*8..16+hf*8+7
      bf16x16 bfr[4];
#pragma unroll
      for (int j = 0; j < 4; ++j) {
        int cpos = wp * 64 + j * 16 + lm;
        union { bf16x16 v; bf16x8 h[2]; } u;
        u.h[0] = *(const bf16x8*)&Bs[b][cpos * LP + ks * 32 + hf * 8];
        u.h[1] = *(const bf16x8*)&Bs[b][cpos * LP + ks * 32 + hf * 8 + 16];
        bfr[j] = u.v;
      }
#pragma unroll
      for (int i = 0; i < 2; ++i)
#pragma unroll
        for (int j = 0; j < 4; ++j)
          acc[i][j] = __builtin_amdgcn_wmma_f32_16x16x32_bf16(
              false, afr[i], false, bfr[j], (short)0, acc[i][j], false, false);
    }
    __syncthreads();                 // all waves done reading buf b before refill
  }

  // epilogue: +bias, ReLU, scatter to (N, Co, T, W) layout
#pragma unroll
  for (int i = 0; i < 2; ++i) {
    int mb = m0 + wm * 32 + i * 16 + hf * 8;   // D layout: VGPR r -> M=r (+8 hi lanes)
#pragma unroll
    for (int j = 0; j < 4; ++j) {
      int pcol = p0 + wp * 64 + j * 16 + lm;   // D layout: N = lane % 16
      int n = pcol >> 7, t = pcol & (TT - 1);
      size_t obase = (size_t)n * (CO * TT * WW_) + (size_t)t * WW_;
#pragma unroll
      for (int r = 0; r < 8; ++r) {
        int m = mb + r;
        int c = m / WW_, w = m - c * WW_;
        float val = acc[i][j][r] + bias[m];
        val = val > 0.f ? val : 0.f;
        out[obase + (size_t)c * (TT * WW_) + w] = val;
      }
    }
  }
}

// ---------- host launcher ----------
extern "C" void kernel_launch(void* const* d_in, const int* in_sizes, int n_in,
                              void* d_out, int out_size, void* d_ws, size_t ws_size,
                              hipStream_t stream) {
  const float* x      = (const float*)d_in[0];
  const float* A_adj  = (const float*)d_in[1];
  const float* resg   = (const float*)d_in[2];
  const float* conv_w = (const float*)d_in[3];
  const float* conv_b = (const float*)d_in[4];
  const float* down_w = (const float*)d_in[5];
  const float* down_b = (const float*)d_in[6];
  const float* dbn_g  = (const float*)d_in[7];
  const float* dbn_b  = (const float*)d_in[8];
  const float* dbn_m  = (const float*)d_in[9];
  const float* dbn_v  = (const float*)d_in[10];
  const float* bn_g   = (const float*)d_in[11];
  const float* bn_b   = (const float*)d_in[12];
  const float* bn_m   = (const float*)d_in[13];
  const float* bn_v   = (const float*)d_in[14];
  float* out = (float*)d_out;

  const size_t WW_BYTES = (size_t)MM * KD * sizeof(__bf16);  // 18,022,400
  const size_t XB_BYTES = (size_t)PP * KD * sizeof(__bf16);  // 13,107,200
  __bf16* ww   = (__bf16*)d_ws;
  __bf16* xbuf = (__bf16*)((char*)d_ws + WW_BYTES);
  float*  bias = (float*)((char*)d_ws + WW_BYTES + XB_BYTES);

  prep_ww_kernel<<<(MM * KD + 255) / 256, 256, 0, stream>>>(
      conv_w, down_w, A_adj, resg, dbn_g, dbn_v, bn_g, bn_v, ww);
  prep_bias_kernel<<<(MM + 255) / 256, 256, 0, stream>>>(
      conv_b, down_b, A_adj, resg, dbn_g, dbn_b, dbn_m, dbn_v,
      bn_g, bn_b, bn_m, bn_v, bias);
  prep_x_kernel<<<(PP * KD + 255) / 256, 256, 0, stream>>>(x, xbuf);

  dim3 grid(MM / 128, PP / 128);   // 22 x 16
  gcn_gemm_kernel<<<grid, 256, 0, stream>>>(ww, xbuf, bias, out);
}
